// SeqAttnWeights_15247133901017
// MI455X (gfx1250) — matrix operand order repaired and checked by reading, not compiled, not run on hardware
//
#include <hip/hip_runtime.h>
#include <hip/hip_bf16.h>

#define BS   16
#define LX   1024
#define LY   1024
#define HDIM 512
#define MDIM 512

#define ROWS 64          // rows of y per workgroup (4 M-tiles of 16)
#define MT   4           // M-tiles per workgroup
#define NTPW 4           // N-tiles per wave (32 N-tiles / 8 waves)

typedef __attribute__((ext_vector_type(16))) __bf16 v16bf;
typedef __attribute__((ext_vector_type(8)))  float  v8f;
typedef __attribute__((ext_vector_type(4)))  float  v4f;
typedef __attribute__((ext_vector_type(4)))  unsigned int v4u;

union BV { v4u q[2]; v16bf v; };

__device__ __forceinline__ unsigned short bf16_rne(float f) {
    unsigned u = __float_as_uint(f);
    u += 0x7FFFu + ((u >> 16) & 1u);          // round-to-nearest-even
    return (unsigned short)(u >> 16);
}
__device__ __forceinline__ float bf16_f32(unsigned short h) {
    return __uint_as_float(((unsigned)h) << 16);
}

// ---------------------------------------------------------------------------
// Kernel 0: split W2 (f32, row-major [MDIM][HDIM]) into bf16 hi/lo planes.
// ---------------------------------------------------------------------------
__global__ __launch_bounds__(256)
void split_w2_kernel(const float* __restrict__ W2,
                     unsigned short* __restrict__ hi,
                     unsigned short* __restrict__ lo) {
    for (int idx = blockIdx.x * blockDim.x + threadIdx.x;
         idx < MDIM * HDIM; idx += gridDim.x * blockDim.x) {
        float f = W2[idx];
        unsigned short h = bf16_rne(f);
        hi[idx] = h;
        lo[idx] = bf16_rne(f - bf16_f32(h));
    }
}

// ---------------------------------------------------------------------------
// Kernel 1: sy[b,l] = sum_m relu( y[b,l,:]·W2[m,:] + b2[m] ) * w_o[m]
// 64-row M-tile per WG so each B tile (global) is reused across 4 M-tiles
// from registers: per K-step = 4 global_load_b128 + 16 ds_load_b128 + 12 WMMA.
// bf16x3 split GEMM on v_wmma_f32_16x16x32_bf16 (f32-class accuracy).
// LDS A layout (per M-tile, per K-step): lane = (m&15) + 16*((kk>>3)&1),
// element e = (kk>>4)*8 + (kk&7) -> 32 contiguous bytes per lane.
// ---------------------------------------------------------------------------
__global__ __launch_bounds__(256)
void sy_wmma_kernel(const float* __restrict__ y,
                    const unsigned short* __restrict__ w2hi,
                    const unsigned short* __restrict__ w2lo,
                    const float* __restrict__ b2,
                    const float* __restrict__ w_o,
                    float* __restrict__ sy) {
    __shared__ __align__(32) unsigned short sAhi[MT * 16 * 32 * 16];  // 64 KB
    __shared__ __align__(32) unsigned short sAlo[MT * 16 * 32 * 16];  // 64 KB
    __shared__ float s_sy[ROWS];

    const int wg = blockIdx.x;            // BS*LY/ROWS = 256 work-groups
    const int b  = wg >> 4;               // 1024/64 = 16 row-groups per batch
    const int l0 = (wg & 15) << 6;
    const size_t ybase = ((size_t)b * LY + l0) * HDIM;

    // Phase 1: stage + split + swizzle the 64 x HDIM y tile into LDS
    for (int idx = threadIdx.x; idx < ROWS * HDIM; idx += 256) {
        int m = idx >> 9;                 // HDIM = 512
        int k = idx & (HDIM - 1);
        float f = y[ybase + (size_t)m * HDIM + k];
        unsigned short h = bf16_rne(f);
        unsigned short l = bf16_rne(f - bf16_f32(h));
        int mt    = m >> 4;
        int mr    = m & 15;
        int kstep = k >> 5;
        int kk    = k & 31;
        int lane  = mr + (((kk >> 3) & 1) << 4);
        int e     = ((kk >> 4) << 3) + (kk & 7);
        int off   = ((((mt << 4) + kstep) << 5) + lane) * 16 + e;
        sAhi[off] = h;
        sAlo[off] = l;
    }
    if (threadIdx.x < ROWS) s_sy[threadIdx.x] = 0.0f;
    __syncthreads();

    const int wave = threadIdx.x >> 5;    // 8 waves
    const int lane = threadIdx.x & 31;
    const int nlo  = lane & 15;
    const int g    = lane >> 4;

    float acc[MT][8];
#pragma unroll
    for (int mt = 0; mt < MT; ++mt)
#pragma unroll
        for (int r = 0; r < 8; ++r) acc[mt][r] = 0.0f;

    // Each wave sweeps 4 N-tiles; B tile registers feed all 4 M-tiles.
    for (int ntl = 0; ntl < NTPW; ++ntl) {
        const int nt = wave * NTPW + ntl;
        const int n  = (nt << 4) + nlo;   // global m-index (B column)
        v8f c[MT];
#pragma unroll
        for (int mt = 0; mt < MT; ++mt) c[mt] = (v8f){};

#pragma unroll 2
        for (int ks = 0; ks < 16; ++ks) {
            // B operands: B[k][n] = W2[n][k]; lane reads two runs of 8 bf16, 16 apart
            // (B layout: lanes 0-15 hold K=0..7,16..23, lanes 16-31 hold K=8..15,24..31)
            const int k0 = (ks << 5) + (g << 3);
            BV bh, bl;
            const v4u* ph = (const v4u*)(w2hi + (size_t)n * HDIM + k0);
            const v4u* pl = (const v4u*)(w2lo + (size_t)n * HDIM + k0);
            bh.q[0] = ph[0]; bh.q[1] = ph[2];
            bl.q[0] = pl[0]; bl.q[1] = pl[2];
#pragma unroll
            for (int mt = 0; mt < MT; ++mt) {
                const int abase = ((((mt << 4) + ks) << 5) + lane) << 4;
                const v16bf a_hi = *(const v16bf*)&sAhi[abase];
                const v16bf a_lo = *(const v16bf*)&sAlo[abase];
                // bf16x3 split: hi*hi + hi*lo + lo*hi (drop lo*lo ~ 2^-32)
                c[mt] = __builtin_amdgcn_wmma_f32_16x16x32_bf16(
                            false, a_lo, false, bh.v, (short)0, c[mt], false, false);
                c[mt] = __builtin_amdgcn_wmma_f32_16x16x32_bf16(
                            false, a_hi, false, bl.v, (short)0, c[mt], false, false);
                c[mt] = __builtin_amdgcn_wmma_f32_16x16x32_bf16(
                            false, a_hi, false, bh.v, (short)0, c[mt], false, false);
            }
        }
        // Fused epilogue: relu(D + b2[n]) * w_o[n], reduce over n into rows
        const float bias = b2[n];
        const float wo   = w_o[n];
#pragma unroll
        for (int mt = 0; mt < MT; ++mt)
#pragma unroll
            for (int r = 0; r < 8; ++r) {
                float v = c[mt][r] + bias;
                v = v > 0.0f ? v : 0.0f;
                acc[mt][r] += v * wo;     // row = mt*16 + r + 8*g
            }
    }

    // Butterfly reduce over the 16 lanes of each half-wave (distinct n's)
#pragma unroll
    for (int mt = 0; mt < MT; ++mt)
#pragma unroll
        for (int r = 0; r < 8; ++r) {
            float v = acc[mt][r];
            v += __shfl_xor(v, 1, 32);
            v += __shfl_xor(v, 2, 32);
            v += __shfl_xor(v, 4, 32);
            v += __shfl_xor(v, 8, 32);
            acc[mt][r] = v;
        }
    if (nlo == 0) {
#pragma unroll
        for (int mt = 0; mt < MT; ++mt)
#pragma unroll
            for (int r = 0; r < 8; ++r)
                atomicAdd(&s_sy[(mt << 4) + r + (g << 3)], acc[mt][r]);  // ds_add_f32
    }
    __syncthreads();
    if (threadIdx.x < ROWS)
        sy[(size_t)b * LY + l0 + threadIdx.x] = s_sy[threadIdx.x];
}

// ---------------------------------------------------------------------------
// Kernel 2: per-batch masked softmax row. Masked lanes are exactly 0 in the
// reference (expf(-1e7 - max) underflows to +0), and the sx[i]+b_o shift
// cancels, so one row per batch serves every i.
// ---------------------------------------------------------------------------
__global__ __launch_bounds__(256)
void rowsoftmax_kernel(const float* __restrict__ sy,
                       const unsigned char* __restrict__ y_mask,
                       float* __restrict__ rows) {
    __shared__ float red[256];
    const int b = blockIdx.x;
    const int t = threadIdx.x;

    float sv[4]; bool mk[4];
    float mx = -3.4e38f;
#pragma unroll
    for (int i = 0; i < 4; ++i) {
        int j = t * 4 + i;
        sv[i] = sy[b * LY + j];
        mk[i] = y_mask[b * LY + j] != 0;
        mx = fmaxf(mx, mk[i] ? -3.4e38f : sv[i]);
    }
    red[t] = mx; __syncthreads();
    for (int s = 128; s > 0; s >>= 1) {
        if (t < s) red[t] = fmaxf(red[t], red[t + s]);
        __syncthreads();
    }
    const float M = red[0];
    __syncthreads();
    const bool allmask = (M <= -3.0e38f);   // ref: all -1e7 -> uniform 1/LY

    float e[4]; float sum = 0.0f;
#pragma unroll
    for (int i = 0; i < 4; ++i) {
        e[i] = allmask ? 1.0f : (mk[i] ? 0.0f : expf(sv[i] - M));
        sum += e[i];
    }
    red[t] = sum; __syncthreads();
    for (int s = 128; s > 0; s >>= 1) {
        if (t < s) red[t] += red[t + s];
        __syncthreads();
    }
    const float inv = 1.0f / red[0];
#pragma unroll
    for (int i = 0; i < 4; ++i)
        rows[b * LY + t * 4 + i] = e[i] * inv;
}

// ---------------------------------------------------------------------------
// Kernel 3: broadcast each batch row to all LX output rows. Pure streaming:
// 64 MB of 128-bit non-temporal stores; source rows (64 KB) stay L2-hot.
// ---------------------------------------------------------------------------
__global__ __launch_bounds__(256)
void bcast_kernel(const float* __restrict__ rows, float* __restrict__ out) {
    const int bi = blockIdx.x;          // 0 .. BS*LX-1
    const int b  = bi >> 10;            // LX = 1024
    const v4f* src = (const v4f*)(rows + (size_t)b * LY);
    v4f* dst = (v4f*)(out + (size_t)bi * LY);
    v4f v = src[threadIdx.x];
    __builtin_nontemporal_store(v, &dst[threadIdx.x]);
}

// ---------------------------------------------------------------------------
extern "C" void kernel_launch(void* const* d_in, const int* in_sizes, int n_in,
                              void* d_out, int out_size, void* d_ws, size_t ws_size,
                              hipStream_t stream) {
    // inputs: 0=x 1=y 2=y_mask 3=W1 4=b1 5=W2 6=b2 7=w_o 8=b_o
    // x/W1/b1/b_o are mathematically eliminated (softmax shift invariance).
    const float*         y      = (const float*)d_in[1];
    const unsigned char* y_mask = (const unsigned char*)d_in[2];
    const float*         W2     = (const float*)d_in[5];
    const float*         b2     = (const float*)d_in[6];
    const float*         w_o    = (const float*)d_in[7];
    float* out = (float*)d_out;

    char* ws = (char*)d_ws;
    float*          sy    = (float*)ws;                           //  64 KB
    float*          rows  = (float*)(ws + (64 << 10));            //  64 KB
    unsigned short* w2hi  = (unsigned short*)(ws + (128 << 10));  // 512 KB
    unsigned short* w2lo  = (unsigned short*)(ws + (640 << 10));  // 512 KB

    split_w2_kernel  <<<256, 256, 0, stream>>>(W2, w2hi, w2lo);
    sy_wmma_kernel   <<<BS * LY / ROWS, 256, 0, stream>>>(y, w2hi, w2lo, b2, w_o, sy);
    rowsoftmax_kernel<<<BS, 256, 0, stream>>>(sy, y_mask, rows);
    bcast_kernel     <<<BS * LX, 256, 0, stream>>>(rows, out);
}
